// DCNBottleneck_5248450036364
// MI455X (gfx1250) — matrix operand order repaired and checked by reading, not compiled, not run on hardware
//
#include <hip/hip_runtime.h>
#include <hip/hip_bf16.h>
#include <math.h>

// ---------------------------------------------------------------------------
// DCN bottleneck for MI455X (gfx1250, wave32):
//   off = conv3x3(x, offset_w) + offset_b          (implicit-GEMM, bf16 WMMA)
//   vals = bilinear deform-sample(x, off)          (branch-free gathers -> LDS)
//   y    = einsum(vals, dcn_w)                     (implicit-GEMM, bf16 WMMA)
//   out  = x + SiLU(BN(y))                         (fused epilogue)
// ---------------------------------------------------------------------------

typedef __bf16 bf16_t;
typedef __attribute__((ext_vector_type(16))) __bf16 v16bf;
typedef __attribute__((ext_vector_type(8)))  __bf16 v8bf;
typedef __attribute__((ext_vector_type(4)))  __bf16 v4bf;
typedef __attribute__((ext_vector_type(8)))  float  v8f;

#define B_    4
#define C_    256
#define H_    80
#define W_    80
#define TAPS  9
#define NOFF  18
#define PLANE (H_ * W_)
#define NPIX  (B_ * PLANE)      // 25600

// LDS A-tile row stride in halves: 40 halves = 80B (16B aligned rows)
#define AST 40

static __device__ __forceinline__ v16bf cat8(v8bf lo, v8bf hi) {
  return __builtin_shufflevector(lo, hi, 0,1,2,3,4,5,6,7,8,9,10,11,12,13,14,15);
}

// --------------------------- weight packing --------------------------------
// Packed layout: [tap][cblk(8)][khalf(2)][n][ci(16)]  (bf16)
// => B fragment for lane L: n = L%16, khalf = L/16, 16 contiguous halves (32B).

__global__ __launch_bounds__(256)
void pack_offw_kernel(const float* __restrict__ w, bf16_t* __restrict__ o) {
  int i = blockIdx.x * 256 + threadIdx.x;            // 9*8*2*32*16 = 73728
  if (i >= 9 * 8 * 2 * 32 * 16) return;
  int ci  = i & 15;
  int n   = (i >> 4) & 31;
  int kh  = (i >> 9) & 1;
  int cb  = (i >> 10) & 7;
  int tap = i >> 13;
  int c = cb * 32 + kh * 16 + ci;
  float v = (n < NOFF) ? w[(n * C_ + c) * TAPS + tap] : 0.0f;   // OIHW, zero-pad n>=18
  o[i] = (bf16_t)v;
}

__global__ __launch_bounds__(256)
void pack_dcnw_kernel(const float* __restrict__ w, bf16_t* __restrict__ o) {
  int i = blockIdx.x * 256 + threadIdx.x;            // 9*8*2*256*16 = 589824
  if (i >= 9 * 8 * 2 * 256 * 16) return;
  int ci  = i & 15;
  int n   = (i >> 4) & 255;
  int kh  = (i >> 12) & 1;
  int cb  = (i >> 13) & 7;
  int tap = i >> 16;
  int c = cb * 32 + kh * 16 + ci;
  o[i] = (bf16_t)w[(n * C_ + c) * TAPS + tap];
}

// --------------------------- offset conv (WMMA) ----------------------------
// Grid: 1600 flat 16-pixel tiles (16 | PLANE => whole tile in one batch image).
// Block: 64 threads = 2 waves; wave k owns out-channels [k*16,k*16+16) of the
// padded-32 N dim. Double-buffered LDS A-tile: one barrier per K-chunk.

__global__ __launch_bounds__(64)
void offconv_kernel(const float* __restrict__ x, const bf16_t* __restrict__ Wb,
                    const float* __restrict__ bias, float* __restrict__ off) {
  __shared__ __bf16 As[2][16][AST];
  const int t    = threadIdx.x;
  const int lane = t & 31;
  const int wave = t >> 5;
  const int pix  = t & 15;        // staging pixel
  const int cg   = t >> 4;        // 0..3 -> 8 contiguous channels each

  // block-uniform batch index / base plane offset (tile never crosses images)
  const int base = blockIdx.x * 16;
  const int b    = base / PLANE;
  const int rb   = base - b * PLANE;   // base offset within the image plane
  const int rem  = rb + pix;
  const int h    = rem / W_;
  const int w    = rem - h * W_;

  v8f acc = {};
  int p = 0;

  for (int tap = 0; tap < TAPS; ++tap) {
    const int ti = tap / 3, tj = tap % 3;
    const int ys = h - 1 + ti;
    const int xs = w - 1 + tj;
    const bool valid = (ys >= 0) && (ys < H_) && (xs >= 0) && (xs < W_);
    const float msk = valid ? 1.0f : 0.0f;
    const int cy = min(max(ys, 0), H_ - 1);
    const int cx = min(max(xs, 0), W_ - 1);
    const int pos = cy * W_ + cx;           // always-safe address
    const float* xb = x + b * C_ * PLANE + pos;

    for (int cb = 0; cb < 8; ++cb) {
      // stage A: 16 px x 32 ch, branch-free, packed 4x-bf16 LDS stores
      #pragma unroll
      for (int jj = 0; jj < 2; ++jj) {
        const int cl = cg * 8 + jj * 4;     // contiguous 4 channels
        v4bf vv;
        #pragma unroll
        for (int j = 0; j < 4; ++j)
          vv[j] = (bf16_t)(xb[(cb * 32 + cl + j) * PLANE] * msk);
        *(v4bf*)&As[p][pix][cl] = vv;
      }
      __syncthreads();
      const int m   = lane & 15;
      const int klo = (lane < 16) ? 0 : 16;
      v16bf afrag = cat8(*(const v8bf*)&As[p][m][klo],
                         *(const v8bf*)&As[p][m][klo + 8]);
      const int khalf = lane >> 4;
      const int n     = wave * 16 + (lane & 15);
      const bf16_t* wp = Wb + ((((tap * 8 + cb) * 2 + khalf) * 32 + n) << 4);
      v16bf bfrag = *(const v16bf*)wp;
      acc = __builtin_amdgcn_wmma_f32_16x16x32_bf16(false, afrag, false, bfrag,
                                                    (short)0, acc, false, false);
      p ^= 1;   // double buffer: no trailing barrier needed
    }
  }
  const int n = wave * 16 + (lane & 15);
  if (n < NOFF) {
    const float bn = bias[n];
    float* op = off + (b * NOFF + n) * PLANE + rb + ((lane >> 4) << 3);
    #pragma unroll
    for (int r = 0; r < 8; ++r)
      op[r] = acc[r] + bn;                   // D row = r + 8*(lane/16)
  }
}

// -------------------- deformable conv + BN + SiLU + residual ---------------
// Grid: 800 flat 32-pixel tiles (32 | PLANE). Block: 256 threads = 8 waves;
// wave k owns columns [k*32,k*32+32) for BOTH 16-row M-subtiles -> 4 WMMAs
// per chunk, B fragments reused. Double-buffered LDS, 1 barrier per chunk.

__global__ __launch_bounds__(256)
void deform_kernel(const float* __restrict__ x, const float* __restrict__ off,
                   const bf16_t* __restrict__ Wd,
                   const float* __restrict__ gamma, const float* __restrict__ beta,
                   const float* __restrict__ mean,  const float* __restrict__ var,
                   float* __restrict__ out) {
  __shared__ __bf16 As[2][32][AST];
  const int t    = threadIdx.x;
  const int lane = t & 31;
  const int wave = t >> 5;
  const int pix  = t & 31;        // staging pixel (0..31)
  const int cg   = t >> 5;        // 0..7 -> 4 contiguous channels each

  // block-uniform batch index / base plane offset
  const int base = blockIdx.x * 32;
  const int b    = base / PLANE;
  const int rb   = base - b * PLANE;
  const int rem  = rb + pix;
  const int h    = rem / W_;
  const int w    = rem - h * W_;

  v8f acc00 = {}, acc01 = {}, acc10 = {}, acc11 = {};
  int p = 0;

  for (int tap = 0; tap < TAPS; ++tap) {
    const int ti = tap / 3, tj = tap % 3;
    // data-dependent bilinear sample position for this pixel & tap
    const float dy = off[(b * NOFF + (tap * 2 + 0)) * PLANE + rem];
    const float dx = off[(b * NOFF + (tap * 2 + 1)) * PLANE + rem];
    const float pyf = dy + (float)(h - 1 + ti);
    const float pxf = dx + (float)(w - 1 + tj);
    const float y0f = floorf(pyf);
    const float x0f = floorf(pxf);
    const int iy0 = (int)y0f, ix0 = (int)x0f;
    const float wy = pyf - y0f, wx = pxf - x0f;
    const bool vy0 = (iy0 >= 0) && (iy0 < H_);
    const bool vy1 = (iy0 + 1 >= 0) && (iy0 + 1 < H_);
    const bool vx0 = (ix0 >= 0) && (ix0 < W_);
    const bool vx1 = (ix0 + 1 >= 0) && (ix0 + 1 < W_);
    const float w00 = (vy0 && vx0) ? (1.f - wy) * (1.f - wx) : 0.f;
    const float w01 = (vy0 && vx1) ? (1.f - wy) * wx         : 0.f;
    const float w10 = (vy1 && vx0) ? wy * (1.f - wx)         : 0.f;
    const float w11 = (vy1 && vx1) ? wy * wx                 : 0.f;
    // clamped (always-safe) corner positions; zero weight kills OOB terms
    const int cy0 = min(max(iy0, 0), H_ - 1), cy1 = min(max(iy0 + 1, 0), H_ - 1);
    const int cx0 = min(max(ix0, 0), W_ - 1), cx1 = min(max(ix0 + 1, 0), W_ - 1);
    const int p00 = cy0 * W_ + cx0, p01 = cy0 * W_ + cx1;
    const int p10 = cy1 * W_ + cx0, p11 = cy1 * W_ + cx1;
    const float* xb = x + b * C_ * PLANE;

    for (int cb = 0; cb < 8; ++cb) {
      // ---- stage A: 32 px x 32 ch bilinear, branch-free, one b64 LDS store
      {
        const int cl = cg * 4;
        v4bf vv;
        #pragma unroll
        for (int j = 0; j < 4; ++j) {
          const float* xp = xb + (cb * 32 + cl + j) * PLANE;
          vv[j] = (bf16_t)(w00 * xp[p00] + w01 * xp[p01] +
                           w10 * xp[p10] + w11 * xp[p11]);
        }
        *(v4bf*)&As[p][pix][cl] = vv;
      }
      __syncthreads();
      // ---- fragments: 2 A (M-subtiles) x 2 B (N-subtiles) -> 4 WMMAs
      const int mr  = lane & 15;
      const int klo = (lane < 16) ? 0 : 16;
      v16bf a0 = cat8(*(const v8bf*)&As[p][mr][klo],
                      *(const v8bf*)&As[p][mr][klo + 8]);
      v16bf a1 = cat8(*(const v8bf*)&As[p][16 + mr][klo],
                      *(const v8bf*)&As[p][16 + mr][klo + 8]);
      const int khalf = lane >> 4;
      const int ncol  = lane & 15;
      const bf16_t* wp = Wd + (((tap * 8 + cb) * 2 + khalf) << 12);  // *256*16
      v16bf bf0 = *(const v16bf*)(wp + ((wave * 32 + ncol) << 4));
      v16bf bf1 = *(const v16bf*)(wp + ((wave * 32 + 16 + ncol) << 4));
      if (cb < 7)  // pull next chunk's weights toward L1
        __builtin_prefetch(wp + ((2 * 256 + wave * 32 + ncol) << 4), 0, 1);
      acc00 = __builtin_amdgcn_wmma_f32_16x16x32_bf16(false, a0, false, bf0,
                                                      (short)0, acc00, false, false);
      acc01 = __builtin_amdgcn_wmma_f32_16x16x32_bf16(false, a0, false, bf1,
                                                      (short)0, acc01, false, false);
      acc10 = __builtin_amdgcn_wmma_f32_16x16x32_bf16(false, a1, false, bf0,
                                                      (short)0, acc10, false, false);
      acc11 = __builtin_amdgcn_wmma_f32_16x16x32_bf16(false, a1, false, bf1,
                                                      (short)0, acc11, false, false);
      p ^= 1;   // double buffer: no trailing barrier
    }
  }

  // ---- fused epilogue: eval-BN + SiLU + residual, NCHW f32 store
  #pragma unroll
  for (int st = 0; st < 2; ++st) {
    const int n = wave * 32 + st * 16 + (lane & 15);
    const float sc = gamma[n] * rsqrtf(var[n] + 1e-5f);
    const float mu = mean[n];
    const float bt = beta[n];
    const int rowbase = (b * C_ + n) * PLANE + rb + ((lane >> 4) << 3);
    #pragma unroll
    for (int mi = 0; mi < 2; ++mi) {
      const v8f a = mi ? (st ? acc11 : acc10) : (st ? acc01 : acc00);
      #pragma unroll
      for (int r = 0; r < 8; ++r) {
        const int idx = rowbase + mi * 16 + r;
        float yv = a[r];
        yv = (yv - mu) * sc + bt;
        const float sig = 1.0f / (1.0f + __expf(-yv));
        out[idx] = x[idx] + yv * sig;
      }
    }
  }
}

// --------------------------------- launch ----------------------------------

extern "C" void kernel_launch(void* const* d_in, const int* in_sizes, int n_in,
                              void* d_out, int out_size, void* d_ws, size_t ws_size,
                              hipStream_t stream) {
  const float* x        = (const float*)d_in[0];
  const float* offset_w = (const float*)d_in[1];
  const float* offset_b = (const float*)d_in[2];
  const float* dcn_w    = (const float*)d_in[3];
  const float* gamma    = (const float*)d_in[4];
  const float* beta     = (const float*)d_in[5];
  const float* mean     = (const float*)d_in[6];
  const float* var      = (const float*)d_in[7];
  float* out = (float*)d_out;

  // workspace carve-up (all 256B aligned):
  //   off : 4*18*80*80 f32      = 1,843,200 B
  //   Wb  : 9*8*2*32*16  bf16   =   147,456 B
  //   Wd  : 9*8*2*256*16 bf16   = 1,179,648 B
  char* ws = (char*)d_ws;
  float*  off = (float*)ws;
  bf16_t* Wb  = (bf16_t*)(ws + 1843200);
  bf16_t* Wd  = (bf16_t*)(ws + 1843200 + 147456);

  pack_offw_kernel<<<288, 256, 0, stream>>>(offset_w, Wb);
  pack_dcnw_kernel<<<2304, 256, 0, stream>>>(dcn_w, Wd);
  offconv_kernel<<<NPIX / 16, 64, 0, stream>>>(x, Wb, offset_b, off);
  deform_kernel<<<NPIX / 32, 256, 0, stream>>>(x, off, Wd, gamma, beta, mean, var, out);
}
